// splanner_26955214750316
// MI455X (gfx1250) — matrix operand order repaired and checked by reading, not compiled, not run on hardware
//
#include <hip/hip_runtime.h>
#include <hip/hip_bf16.h>

// ---------------------------------------------------------------------------
// Problem constants
#define ASZ 50
#define BB  64
#define KK  512
#define TT  256
#define HSZ 1024
#define NK  515          // 3 + K
#define NKP 528          // padded to multiple of 16
#define DP  64           // ASZ padded to 64 (two bf16 WMMA k-chunks)

typedef __attribute__((ext_vector_type(16))) __bf16          v16bf;
typedef __attribute__((ext_vector_type(16))) unsigned short  v16us;
typedef __attribute__((ext_vector_type(8)))  float           v8f;

// Truncation-based split: a ~= hi + lo with ~2^-15 relative error.
// hi = top 16 bits of f32; residual (exact in f32) truncated to bf16.
__device__ __forceinline__ void split2(float a, unsigned short& h, unsigned short& l) {
    unsigned int u = __float_as_uint(a);
    h = (unsigned short)(u >> 16);
    float hf = __uint_as_float(u & 0xFFFF0000u);
    l = (unsigned short)(__float_as_uint(a - hf) >> 16);
}

__device__ __forceinline__ void split16(const float* x, v16bf& hi, v16bf& lo) {
    v16us vh, vl;
#pragma unroll
    for (int i = 0; i < 16; ++i) {
        unsigned short h, l;
        split2(x[i], h, l);
        vh[i] = h; vl[i] = l;
    }
    hi = __builtin_bit_cast(v16bf, vh);
    lo = __builtin_bit_cast(v16bf, vl);
}

__device__ __forceinline__ v16bf ldbf16(const unsigned short* p) {
    return __builtin_bit_cast(v16bf, *(const v16us*)p);
}

__device__ __forceinline__ v8f wmma_bf16(v16bf a, v16bf b, v8f c) {
    return __builtin_amdgcn_wmma_f32_16x16x32_bf16(
        false, a, false, b, (short)0, c, false, false);
}

__device__ __forceinline__ float sigmoidf_(float x) {
    return 1.f / (1.f + __expf(-x));
}

// ---------------------------------------------------------------------------
// Kernel 1: prep. Per-batch block (64 blocks x 256 thr):
//   h0 = hx @ clin_w.T + clin_b (padded to 64)
//   AKhi/AKlo rows 0..2 = split(emb) (padded), rows 515..527 zeroed
//   Bhi/Blo = split(klin_w) padded to 64 rows (block b owns row n=b)
//   gi[0] = emb_row1 @ w_ih.T + b_ih,  gi[1] = emb_row0 @ w_ih.T + b_ih
//   firstmask[b][k] = first step s whose goldup == k (else big)
__global__ __launch_bounds__(256) void prep_kernel(
    const float* __restrict__ hx, const int* __restrict__ entlens,
    const int* __restrict__ gold, const float* __restrict__ emb,
    const float* __restrict__ w_ih, const float* __restrict__ b_ih,
    const float* __restrict__ clin_w, const float* __restrict__ clin_b,
    const float* __restrict__ klin_w,
    float* __restrict__ h0, float* __restrict__ gi, int* __restrict__ fm,
    unsigned short* __restrict__ Bhi, unsigned short* __restrict__ Blo,
    unsigned short* __restrict__ AKhi, unsigned short* __restrict__ AKlo)
{
    const int b = blockIdx.x, tid = threadIdx.x;

    // h0 (padded to 64)
    if (tid < 64) {
        float v = 0.f;
        if (tid < ASZ) {
            const float* xr = hx + (size_t)b * HSZ;
            const float* wr = clin_w + (size_t)tid * HSZ;
            float acc = clin_b[tid];
            for (int d = 0; d < HSZ; d += 4)
                acc += xr[d]*wr[d] + xr[d+1]*wr[d+1] + xr[d+2]*wr[d+2] + xr[d+3]*wr[d+3];
            v = acc;
        }
        h0[b * 64 + tid] = v;
    }

    // B operand (klin_w) split to bf16 hi/lo, row n = b (zeros for n >= 50)
    {
        const bool nv = b < ASZ;
        for (int k = tid; k < HSZ; k += 256) {
            float v = nv ? klin_w[(size_t)b * HSZ + k] : 0.f;
            unsigned short h, l;
            split2(v, h, l);
            Bhi[(size_t)b * HSZ + k] = h;
            Blo[(size_t)b * HSZ + k] = l;
        }
    }

    // emb rows 0..2 of allkeys (bf16 hi/lo form)
    if (tid < 3 * 64) {
        int rr = tid >> 6, d = tid & 63;
        float v = (d < ASZ) ? emb[rr * ASZ + d] : 0.f;
        unsigned short h, l;
        split2(v, h, l);
        size_t idx = ((size_t)b * NKP + rr) * DP + d;
        AKhi[idx] = h; AKlo[idx] = l;
    }
    // zero pad rows 515..527
    for (int idx = tid; idx < (NKP - NK) * 64; idx += 256) {
        int rr = NK + idx / 64, d = idx % 64;
        size_t ix = ((size_t)b * NKP + rr) * DP + d;
        AKhi[ix] = 0; AKlo[ix] = 0;
    }

    // gi: two constant 150-vectors (block 0 only)
    if (b == 0 && tid < 3 * ASZ) {
        const float* wr = w_ih + tid * ASZ;
        float a0 = b_ih[tid], a1 = b_ih[tid];
        for (int d = 0; d < ASZ; ++d) {
            a0 += emb[ASZ + d] * wr[d];   // prev0  = emb row 1
            a1 += emb[d] * wr[d];         // prev_* = emb row 0
        }
        gi[tid] = a0; gi[160 + tid] = a1;
    }

    // firstmask
    for (int k = tid; k < NKP; k += 256) fm[b * NKP + k] = 0x3FFFFFFF;
    __syncthreads();
    if (tid == 0) {
        int el = entlens[b];
        for (int s = TT - 1; s >= 0; --s) {   // descending => final = min s
            int g = gold[b * TT + s];
            if (g < 3) g = 0;
            if (s == el) g = 0;
            fm[b * NKP + g] = s;
        }
    }
}

// ---------------------------------------------------------------------------
// Kernel 2: kproj = keys(32768x1024) @ klin_w.T + klin_b.
// One wave owns a 16-row M-tile and ALL four 16-col N-tiles (N=64 padded):
// A is split to bf16 hi/lo ONCE per K-chunk and feeds 12 WMMAs; B comes
// pre-split from memory (no conversion VALU). Output stored directly in
// bf16 hi/lo form for the attention kernel. HBM-bound (~134 MB of keys).
__global__ __launch_bounds__(256) void kproj_wmma_kernel(
    const float* __restrict__ keys,
    const unsigned short* __restrict__ Bhi, const unsigned short* __restrict__ Blo,
    const float* __restrict__ klin_b,
    unsigned short* __restrict__ AKhi, unsigned short* __restrict__ AKlo)
{
    const int tid = threadIdx.x;
    const int lane = tid & 31, w = tid >> 5;
    const int half = lane >> 4, r = lane & 15;
    const int mBase = blockIdx.x * 128 + w * 16;        // global row (b*512 + k)
    const int b = mBase >> 9;
    const int kTileLoc = mBase & 511;

    const float* Arow = keys + (size_t)(mBase + r) * HSZ;   // A row M = lane&15

    v8f acc[4] = {{}, {}, {}, {}};
    for (int kt = 0; kt < HSZ / 32; ++kt) {
        const int kb = kt * 32;
        __builtin_prefetch(Arow + kb + 64, 0, 1);

        // A 16x32 bf16 layout: lane half 0: K = kb+0..15; half 1: +16
        float av[16] __attribute__((aligned(16)));
        const int a0 = kb + half * 16;
        *(float4*)(av + 0)  = *(const float4*)(Arow + a0);
        *(float4*)(av + 4)  = *(const float4*)(Arow + a0 + 4);
        *(float4*)(av + 8)  = *(const float4*)(Arow + a0 + 8);
        *(float4*)(av + 12) = *(const float4*)(Arow + a0 + 12);
        v16bf ahi, alo;
        split16(av, ahi, alo);

        // B 32x16 bf16 layout: lane col N = lane&15; half 0: K=kb+0..15, half 1: +16
        const int boff = kb + half * 16;
#pragma unroll
        for (int nt = 0; nt < 4; ++nt) {
            const size_t brow = (size_t)(nt * 16 + r) * HSZ + boff;
            v16bf bhi = ldbf16(Bhi + brow);
            v16bf blo = ldbf16(Blo + brow);
            acc[nt] = wmma_bf16(alo, bhi, acc[nt]);
            acc[nt] = wmma_bf16(ahi, blo, acc[nt]);
            acc[nt] = wmma_bf16(ahi, bhi, acc[nt]);
        }
    }

    // Store D (+bias) in split-bf16 form; D row M = rr + 8*half, col = nt*16 + r
#pragma unroll
    for (int nt = 0; nt < 4; ++nt) {
        const int ncol = nt * 16 + r;
        const float bias = (ncol < ASZ) ? klin_b[ncol] : 0.f;
        const size_t base = ((size_t)b * NKP + 3 + kTileLoc) * DP + ncol;
#pragma unroll
        for (int rr = 0; rr < 8; ++rr) {
            const int M = rr + 8 * half;
            unsigned short h, l;
            split2(acc[nt][rr] + bias, h, l);
            AKhi[base + (size_t)M * DP] = h;
            AKlo[base + (size_t)M * DP] = l;
        }
    }
}

// ---------------------------------------------------------------------------
// Kernel 3: sequential GRU per batch (64 indep. chains). gi is two constant
// 150-vectors. Writes H[b][t][0..63] (cols 50..63 zero for WMMA K-padding).
__global__ __launch_bounds__(192) void gru_kernel(
    const float* __restrict__ w_hh, const float* __restrict__ b_hh,
    const float* __restrict__ h0, const float* __restrict__ gi,
    float* __restrict__ H)
{
    const int b = blockIdx.x, j = threadIdx.x;
    __shared__ float h_sh[2][64];
    __shared__ float gs[3 * ASZ + 2];

    float whh[ASZ] = {0.f};
    float bh = 0.f, g0 = 0.f, g1 = 0.f, gn0 = 0.f, gn1 = 0.f;
    if (j < 3 * ASZ) {
#pragma unroll
        for (int d = 0; d < ASZ; ++d) whh[d] = w_hh[j * ASZ + d];
        bh = b_hh[j];
        g0 = gi[j]; g1 = gi[160 + j];
    }
    if (j < ASZ) { gn0 = gi[2 * ASZ + j]; gn1 = gi[160 + 2 * ASZ + j]; }
    if (j < 64)  { h_sh[0][j] = h0[b * 64 + j]; h_sh[1][j] = 0.f; }
    __syncthreads();

    int cur = 0;
    for (int t = 0; t < TT; ++t) {
        if (j < 3 * ASZ) {
            float acc = bh;
#pragma unroll
            for (int d = 0; d < ASZ; ++d) acc += whh[d] * h_sh[cur][d];
            if (j < 2 * ASZ) acc += (t == 0 ? g0 : g1);   // r,z gates: gi + gh
            gs[j] = acc;                                   // n gate: gh only
        }
        __syncthreads();
        const int nxt = cur ^ 1;
        if (j < ASZ) {
            float rr = sigmoidf_(gs[j]);
            float zz = sigmoidf_(gs[ASZ + j]);
            float gin = (t == 0 ? gn0 : gn1);
            float nn = tanhf(gin + rr * gs[2 * ASZ + j]);
            float hn = (1.f - zz) * nn + zz * h_sh[cur][j];
            h_sh[nxt][j] = hn;
            H[((size_t)b * TT + t) * DP + j] = hn;
        } else if (j >= 150 && j < 164) {
            H[((size_t)b * TT + t) * DP + ASZ + (j - 150)] = 0.f;  // K pad
        }
        __syncthreads();
        cur = nxt;
    }
}

// ---------------------------------------------------------------------------
// Kernel 4: per (b, 16-row t-tile): scores via split-bf16 WMMA into LDS,
// then fused masked softmax (mask = m0 OR firstmask<t) and output write.
// B operand (allkeys) is pre-split in memory; only the tiny H tile is split.
__global__ __launch_bounds__(256) void attn_softmax_kernel(
    const unsigned short* __restrict__ AKhi, const unsigned short* __restrict__ AKlo,
    const float* __restrict__ H,
    const unsigned char* __restrict__ maskIn, const int* __restrict__ fm,
    float* __restrict__ out)
{
    const int bid = blockIdx.x;
    const int b = bid >> 4, t0 = (bid & 15) * 16;
    const int tid = threadIdx.x, lane = tid & 31, w = tid >> 5;
    const int half = lane >> 4, r = lane & 15;

    __shared__ float sc[16][532];   // stride 532 avoids half-wave bank aliasing

    // Hoist A (H tile, 16x64) as hi/lo bf16 for both K-chunks
    const float* Hrow = H + ((size_t)b * TT + t0 + r) * DP;
    v16bf ah[2], al[2];
#pragma unroll
    for (int kt = 0; kt < 2; ++kt) {
        float av[16] __attribute__((aligned(16)));
        const int a0 = kt * 32 + half * 16;
        *(float4*)(av + 0)  = *(const float4*)(Hrow + a0);
        *(float4*)(av + 4)  = *(const float4*)(Hrow + a0 + 4);
        *(float4*)(av + 8)  = *(const float4*)(Hrow + a0 + 8);
        *(float4*)(av + 12) = *(const float4*)(Hrow + a0 + 12);
        split16(av, ah[kt], al[kt]);
    }

    for (int nt = w; nt < NKP / 16; nt += 8) {          // wave-uniform trip count
        const int n = nt * 16 + r;
        const size_t brow = ((size_t)b * NKP + n) * DP;
        v8f acc = {};
#pragma unroll
        for (int kt = 0; kt < 2; ++kt) {
            const size_t off = brow + kt * 32 + half * 16;
            v16bf bhi = ldbf16(AKhi + off);
            v16bf blo = ldbf16(AKlo + off);
            acc = wmma_bf16(al[kt], bhi, acc);
            acc = wmma_bf16(ah[kt], blo, acc);
            acc = wmma_bf16(ah[kt], bhi, acc);
        }
#pragma unroll
        for (int rr = 0; rr < 8; ++rr) sc[rr + 8 * half][n] = acc[rr];
    }
    __syncthreads();

    // Softmax: wave w owns rows 2w, 2w+1
    for (int ri = 0; ri < 2; ++ri) {
        const int tl = w * 2 + ri, t = t0 + tl;
        float mx = -1e30f;
        for (int k = lane; k < NK; k += 32) {
            bool msk = (k >= 3 && maskIn[(size_t)b * KK + (k - 3)] != 0) ||
                       (fm[b * NKP + k] < t);
            if (!msk) mx = fmaxf(mx, sc[tl][k]);
        }
#pragma unroll
        for (int off = 16; off > 0; off >>= 1) mx = fmaxf(mx, __shfl_xor(mx, off));

        float sum = 0.f;
        for (int k = lane; k < NK; k += 32) {
            bool msk = (k >= 3 && maskIn[(size_t)b * KK + (k - 3)] != 0) ||
                       (fm[b * NKP + k] < t);
            float e = msk ? 0.f : __expf(sc[tl][k] - mx);
            sc[tl][k] = e;
            sum += e;
        }
#pragma unroll
        for (int off = 16; off > 0; off >>= 1) sum += __shfl_xor(sum, off);
        const float inv = 1.f / sum;

        for (int k = lane; k < NK; k += 32)
            out[((size_t)b * TT + t) * NK + k] = sc[tl][k] * inv;
    }
}

// ---------------------------------------------------------------------------
extern "C" void kernel_launch(void* const* d_in, const int* in_sizes, int n_in,
                              void* d_out, int out_size, void* d_ws, size_t ws_size,
                              hipStream_t stream) {
    const float*         hx      = (const float*)d_in[0];
    const float*         keys    = (const float*)d_in[1];
    const unsigned char* mask    = (const unsigned char*)d_in[2];
    const int*           entlens = (const int*)d_in[3];
    const int*           gold    = (const int*)d_in[4];
    const float*         emb     = (const float*)d_in[5];
    const float*         w_ih    = (const float*)d_in[6];
    const float*         w_hh    = (const float*)d_in[7];
    const float*         b_ih    = (const float*)d_in[8];
    const float*         b_hh    = (const float*)d_in[9];
    const float*         clin_w  = (const float*)d_in[10];
    const float*         clin_b  = (const float*)d_in[11];
    const float*         klin_w  = (const float*)d_in[12];
    const float*         klin_b  = (const float*)d_in[13];

    // Workspace layout (float-indexed; every region 32B-aligned)
    float* ws = (float*)d_ws;
    float* H   = ws;                                     // 64*256*64 = 1,048,576 f
    float* h0  = H + (size_t)BB * TT * DP;               // 4,096 f
    float* gi  = h0 + BB * 64;                           // 320 f
    int*   fm  = (int*)(gi + 2 * 160);                   // 64*528 = 33,792 i
    unsigned short* Bhi  = (unsigned short*)(ws + 1086784);   // 64*1024 us
    unsigned short* Blo  = (unsigned short*)(ws + 1119552);   // 64*1024 us
    unsigned short* AKhi = (unsigned short*)(ws + 1152320);   // 64*528*64 us
    unsigned short* AKlo = (unsigned short*)(ws + 2233664);   // 64*528*64 us
    // total ~3.32M floats ~= 13.3 MB

    prep_kernel<<<BB, 256, 0, stream>>>(hx, entlens, gold, emb, w_ih, b_ih,
                                        clin_w, clin_b, klin_w,
                                        h0, gi, fm, Bhi, Blo, AKhi, AKlo);
    kproj_wmma_kernel<<<(BB * KK) / 128, 256, 0, stream>>>(keys, Bhi, Blo, klin_b,
                                                           AKhi, AKlo);
    gru_kernel<<<BB, 192, 0, stream>>>(w_hh, b_hh, h0, gi, H);
    attn_softmax_kernel<<<BB * (TT / 16), 256, 0, stream>>>(AKhi, AKlo, H, mask, fm,
                                                            (float*)d_out);
}
